// GCNBlock_53060025974955
// MI455X (gfx1250) — matrix operand order repaired and checked by reading, not compiled, not run on hardware
//
#include <hip/hip_runtime.h>
#include <hip/hip_bf16.h>

// ---------------------------------------------------------------------------
// GCN block for MI455X (gfx1250, wave32, WMMA).
// Layout everywhere: (B, N, T, C) fp32  -> flat row r = (b*N + n)*T + t, 64 ch.
// ---------------------------------------------------------------------------

typedef __attribute__((ext_vector_type(16))) __bf16 v16bf;
typedef __attribute__((ext_vector_type(8)))  float  v8f;

#define GCN_N  10000
#define GCN_B  2
#define GCN_T  12
#define GCN_C  64
#define SLAB   (GCN_T * GCN_C)        // 768 floats per (b, n)
#define SLAB4  (SLAB / 4)             // 192 float4 per (b, n)

// Native L2 f32 atomic add (no-return form -> STOREcnt; drained by S_ENDPGM's
// implicit wait-idle). Device scope so the RMW is visible across all WGPs.
__device__ __forceinline__ void atomic_add_f32(float* p, float v) {
  asm volatile("global_atomic_add_f32 %0, %1, off scope:SCOPE_DEV"
               :
               : "v"(p), "v"(v)
               : "memory");
}

__device__ __forceinline__ v8f wmma_bf16(v16bf a, v16bf b, v8f c) {
  // (neg_a, A, neg_b, B, c_mod, C, reuse_a, reuse_b)
  return __builtin_amdgcn_wmma_f32_16x16x32_bf16(false, a, false, b, (short)0, c,
                                                 false, false);
}

// --------------------------- degree / norm ---------------------------------
__global__ void deg_init(float* __restrict__ deg, int n) {
  int i = blockIdx.x * blockDim.x + threadIdx.x;
  if (i < n) deg[i] = 1.0f;  // self loop
}

__global__ void deg_count(const int* __restrict__ dst, float* __restrict__ deg, int E) {
  int i = blockIdx.x * blockDim.x + threadIdx.x;
  if (i < E) atomic_add_f32(&deg[dst[i]], 1.0f);
}

__global__ void deg_to_dinv(float* __restrict__ deg, int n) {
  int i = blockIdx.x * blockDim.x + threadIdx.x;
  if (i < n) {
    float d = deg[i];
    deg[i] = (d > 0.0f) ? __frsqrt_rn(d) : 0.0f;
  }
}

// --------------------------- WMMA GEMM (rows x 64) @ (64 x 64) -------------
// Each wave computes a 16x64 output tile with 8 v_wmma_f32_16x16x32_bf16.
// RELU_BIAS: A-operand is relu(A + bias[channel]) (fused conv1 activation).
template <int RELU_BIAS>
__global__ __launch_bounds__(256) void gemm64_wmma(
    const float* __restrict__ A, const float* __restrict__ Wm,
    const float* __restrict__ bias, float* __restrict__ out, int numRows) {
  const int lane = threadIdx.x & 31;
  const int wave = threadIdx.x >> 5;
  const int ln = lane & 15;   // column / row-within-tile selector
  const int lh = lane >> 4;   // lane half (0/1)
  const int ko = lh << 3;     // +8 K offset for upper lane half (A operand)

  // ---- B fragments (W is 64x64 row-major, [k][n]); hoisted out of loop ----
  // B layout: lane -> column n = lane%16, element e -> K = e + 8*(e>=8) + 8*(lane>=16).
  v16bf bf[4][2];
#pragma unroll
  for (int j = 0; j < 4; ++j) {
#pragma unroll
    for (int h = 0; h < 2; ++h) {
      const float* wp = Wm + (size_t)(32 * h + 8 * lh) * 64 + 16 * j + ln;
#pragma unroll
      for (int e = 0; e < 16; ++e) bf[j][h][e] = (__bf16)wp[(size_t)(e + ((e >> 3) << 3)) * 64];
    }
  }

  const int numTiles = numRows >> 4;
  const int tileStride = gridDim.x * 8;
  for (int tile = blockIdx.x * 8 + wave; tile < numTiles; tile += tileStride) {
    const int rowBase = tile << 4;
    // A layout: lane -> row m = lane%16; element e -> K = e + 8*(e>=8) + ko.
    const float* rp = A + (size_t)(rowBase + ln) * 64 + ko;
    v16bf a0, a1;
#pragma unroll
    for (int e = 0; e < 8; ++e) {
      float v0 = rp[e];        // K = ko + e
      float v1 = rp[e + 16];   // K = ko + 16 + e
      float v2 = rp[e + 32];   // K = 32 + ko + e
      float v3 = rp[e + 48];   // K = 32 + ko + 16 + e
      if constexpr (RELU_BIAS) {
        v0 = fmaxf(v0 + bias[ko + e], 0.0f);
        v1 = fmaxf(v1 + bias[ko + e + 16], 0.0f);
        v2 = fmaxf(v2 + bias[ko + e + 32], 0.0f);
        v3 = fmaxf(v3 + bias[ko + e + 48], 0.0f);
      }
      a0[e] = (__bf16)v0;
      a0[e + 8] = (__bf16)v1;
      a1[e] = (__bf16)v2;
      a1[e + 8] = (__bf16)v3;
    }

    v8f acc[4] = {};
#pragma unroll
    for (int j = 0; j < 4; ++j) {
      acc[j] = wmma_bf16(a0, bf[j][0], acc[j]);
      acc[j] = wmma_bf16(a1, bf[j][1], acc[j]);
    }

    // D layout: VGPR r -> row m = r + 8*lh, col n = 16*j + ln.
    float* op = out + (size_t)(rowBase + 8 * lh) * 64 + ln;
#pragma unroll
    for (int j = 0; j < 4; ++j) {
#pragma unroll
      for (int r = 0; r < 8; ++r) op[(size_t)r * 64 + 16 * j] = acc[j][r];
    }
  }
}

// --------------------------- aggregation -----------------------------------
// agg[b,n,:,:] = xw[b,n,:,:] * dinv[n]^2   (self-loop term; also zero-inits)
__global__ void self_loop_init(const float* __restrict__ xw,
                               const float* __restrict__ dinv,
                               float* __restrict__ agg, int total4) {
  int i = blockIdx.x * blockDim.x + threadIdx.x;
  if (i >= total4) return;
  size_t flat = (size_t)i * 4;
  int n = (int)((flat / SLAB) % GCN_N);
  float w = dinv[n];
  w *= w;
  float4 v = ((const float4*)xw)[i];
  v.x *= w; v.y *= w; v.z *= w; v.w *= w;
  ((float4*)agg)[i] = v;
}

// One block (384 threads = 12 waves) per edge: moves the full 6KB slab
// (2 batches x 768 floats) with float4 gathers + 4 f32 atomics per thread.
// Both xw and agg are L2-resident (61 MB each vs 192 MB L2), so the atomics
// run at L2 atomic-unit speed.
__global__ __launch_bounds__(384) void edge_scatter(
    const float* __restrict__ xw, const int* __restrict__ src,
    const int* __restrict__ dst, const float* __restrict__ dinv,
    float* __restrict__ agg) {
  const int e = blockIdx.x;
  const int p = threadIdx.x;        // 0..383
  const int s = src[e];
  const int d = dst[e];
  const float w = dinv[s] * dinv[d];
  const int b = p / SLAB4;          // batch 0/1
  const int off = (p % SLAB4) * 4;  // float offset within slab
  const float4 v =
      *(const float4*)(xw + ((size_t)b * GCN_N + s) * SLAB + off);
  float* o = agg + ((size_t)b * GCN_N + d) * SLAB + off;
  atomic_add_f32(o + 0, v.x * w);
  atomic_add_f32(o + 1, v.y * w);
  atomic_add_f32(o + 2, v.z * w);
  atomic_add_f32(o + 3, v.w * w);
}

// --------------------------- epilogue --------------------------------------
__global__ void bias_sigmoid_inplace(float* __restrict__ io,
                                     const float* __restrict__ bias, int total4) {
  int i = blockIdx.x * blockDim.x + threadIdx.x;
  if (i >= total4) return;
  int cb = (i * 4) & 63;  // channel of first element (C=64)
  float4 v = ((float4*)io)[i];
  v.x = 1.0f / (1.0f + __expf(-(v.x + bias[cb + 0])));
  v.y = 1.0f / (1.0f + __expf(-(v.y + bias[cb + 1])));
  v.z = 1.0f / (1.0f + __expf(-(v.z + bias[cb + 2])));
  v.w = 1.0f / (1.0f + __expf(-(v.w + bias[cb + 3])));
  ((float4*)io)[i] = v;
}

// ---------------------------------------------------------------------------
extern "C" void kernel_launch(void* const* d_in, const int* in_sizes, int n_in,
                              void* d_out, int out_size, void* d_ws, size_t ws_size,
                              hipStream_t stream) {
  const float* X  = (const float*)d_in[0];   // (B,N,T,C) fp32
  const int*   ei = (const int*)d_in[1];     // (2,E) int32
  const float* W1 = (const float*)d_in[2];
  const float* b1 = (const float*)d_in[3];
  const float* W2 = (const float*)d_in[4];
  const float* b2 = (const float*)d_in[5];
  float* out = (float*)d_out;

  const int E = in_sizes[1] / 2;
  const int* src = ei;
  const int* dst = ei + E;
  const int numRows = in_sizes[0] / GCN_C;           // B*T*N = 240000
  const int total  = in_sizes[0];                    // 15,360,000 floats
  const int total4 = total / 4;

  // workspace layout
  char* ws = (char*)d_ws;
  float* dinv = (float*)ws;                                   // N floats
  float* xw1  = (float*)(ws + (1 << 16));                     // 61.44 MB
  float* agg1 = xw1 + (size_t)total;                          // 61.44 MB
  float* xw2  = agg1 + (size_t)total;                         // 61.44 MB
  // agg2 lives in d_out

  const int thr = 256;
  const int ewBlocks = (total4 + thr - 1) / thr;

  // normalization coefficients
  deg_init<<<(GCN_N + thr - 1) / thr, thr, 0, stream>>>(dinv, GCN_N);
  deg_count<<<(E + thr - 1) / thr, thr, 0, stream>>>(dst, dinv, E);
  deg_to_dinv<<<(GCN_N + thr - 1) / thr, thr, 0, stream>>>(dinv, GCN_N);

  // conv1: xw1 = X @ W1 ; agg1 = D^-1/2 A D^-1/2 xw1
  gemm64_wmma<0><<<625, 256, 0, stream>>>(X, W1, nullptr, xw1, numRows);
  self_loop_init<<<ewBlocks, thr, 0, stream>>>(xw1, dinv, agg1, total4);
  edge_scatter<<<E, 384, 0, stream>>>(xw1, src, dst, dinv, agg1);

  // conv2: xw2 = relu(agg1 + b1) @ W2 ; out = D^-1/2 A D^-1/2 xw2
  gemm64_wmma<1><<<625, 256, 0, stream>>>(agg1, W2, b1, xw2, numRows);
  self_loop_init<<<ewBlocks, thr, 0, stream>>>(xw2, dinv, out, total4);
  edge_scatter<<<E, 384, 0, stream>>>(xw2, src, dst, dinv, out);

  // out = sigmoid(out + b2)
  bias_sigmoid_inplace<<<ewBlocks, thr, 0, stream>>>(out, b2, total4);
}
